// MDRNN2D_68805376082340
// MI455X (gfx1250) — matrix operand order repaired and checked by reading, not compiled, not run on hardware
//
#include <hip/hip_runtime.h>
#include <math.h>

// MD-LSTM 4-direction diagonal scan for MI455X (gfx1250), wave32 + WMMA +
// async global->LDS double-buffered x staging.
//
// Shapes: x (4,128,128,64) f32, Wk (4,64,256), Wr (4,64,256), b (4,256),
// out (4,128,128,256) f32 (4 directions concatenated on last axis).
//
// 16 persistent workgroups = (dir, batch). Per diagonal d:
//   A = [x_diag (f32->f16 at frag build) | h_prev_avg (f16)]  (128 x 128)
//   WcatT = [Wk; Wr]^T in LDS as f16                          (256 x 128)
//   Z = A @ Wcat + b  via v_wmma_f32_16x16x32_f16 (4 k-steps per tile)
//   lane-local LSTM gating; h/c state in 2-way rotating LDS buffers;
//   x for diagonal d+1 streams in via GLOBAL_LOAD_ASYNC_TO_LDS_B128
//   concurrently with the WMMA/gating of diagonal d.

typedef _Float16 v16h __attribute__((ext_vector_type(16)));
typedef float    v8f  __attribute__((ext_vector_type(8)));
typedef _Float16 h4v  __attribute__((ext_vector_type(4)));
typedef float    f4v  __attribute__((ext_vector_type(4)));
typedef int      v4i_t __attribute__((__vector_size__(16)));

#define H_     128
#define W_     128
#define C_     64
#define U_     64
#define NG     256          // 4*U gate width
#define DMAX   128
#define NDIAG  (H_ + W_ - 1)
#define THREADS 512         // 16 wave32 waves

// LDS partition (bytes)
#define OFF_W   0                       // WcatT f16 [256][128]        65536
#define OFF_B   (OFF_W  + 256*128*2)    // bias f32 [256]               1024
#define OFF_X   (OFF_B  + 256*4)        // x staging f32 x2 [128][64]  65536
#define OFF_HA  (OFF_X  + 2*128*64*4)   // h_prev avg f16 [128][64]    16384
#define OFF_CP  (OFF_HA + 128*64*2)     // c_prev avg f32 [128][64]    32768
#define OFF_H2  (OFF_CP + 128*64*4)     // h state f16 x2 [128][64]    32768
#define OFF_C2  (OFF_H2 + 2*128*64*2)   // c state f32 x2 [128][64]    65536
#define SMEM_BYTES (OFF_C2 + 2*128*64*4)  // = 279552 < 320 KB

#ifndef __has_builtin
#define __has_builtin(x) 0
#endif

#if __has_builtin(__builtin_amdgcn_global_load_async_to_lds_b128)
#define USE_ASYNC_LDS 1
#else
#define USE_ASYNC_LDS 0
#endif

// 16-byte global -> LDS copy. Async (ASYNCcnt-tracked, no VGPR round trip)
// when the gfx1250 builtin is available, otherwise a plain synchronous copy.
__device__ __forceinline__ void copy16_to_lds(const float* g, float* l) {
#if USE_ASYNC_LDS
    __builtin_amdgcn_global_load_async_to_lds_b128(
        (__attribute__((address_space(1))) v4i_t*)(float*)g,
        (__attribute__((address_space(3))) v4i_t*)l,
        0, 0);
#else
    *(float4*)l = *(const float4*)g;
#endif
}

__device__ __forceinline__ void wait_async_copies() {
#if USE_ASYNC_LDS
#if __has_builtin(__builtin_amdgcn_s_wait_asynccnt)
    __builtin_amdgcn_s_wait_asynccnt(0);
#else
    asm volatile("s_wait_asynccnt 0x0" ::: "memory");
#endif
#endif
}

__global__ __launch_bounds__(THREADS)
void mdlstm_scan_kernel(const float* __restrict__ x,
                        const float* __restrict__ Wk,
                        const float* __restrict__ Wr,
                        const float* __restrict__ bvec,
                        float* __restrict__ out)
{
    extern __shared__ char smem[];
    _Float16* sW  = (_Float16*)(smem + OFF_W);   // [n][k] k-contiguous
    float*    sB  = (float*)   (smem + OFF_B);
    float*    sX  = (float*)   (smem + OFF_X);   // 2 x [m][k] f32
    _Float16* sHA = (_Float16*)(smem + OFF_HA);  // [m][k]
    float*    sCP = (float*)   (smem + OFF_CP);  // [m][u]
    _Float16* sH  = (_Float16*)(smem + OFF_H2);  // 2 x [m][u]
    float*    sC  = (float*)   (smem + OFF_C2);  // 2 x [m][u]

    const int dir = blockIdx.x >> 2;     // 0..3
    const int bb  = blockIdx.x & 3;      // batch
    const bool fh = (dir & 2) != 0;      // _FLIPS order: (F,F),(F,T),(T,F),(T,T)
    const bool fw = (dir & 1) != 0;

    const int tid = threadIdx.x;

    // ---- one-time loads: WcatT (f16, [n][k]) and bias; zero state ----
    for (int i = tid; i < NG; i += THREADS) sB[i] = bvec[dir * NG + i];
    for (int i = tid; i < 128 * NG; i += THREADS) {
        int k = i & 127, n = i >> 7;
        float v = (k < 64) ? Wk[((size_t)dir * 64 + k) * NG + n]
                           : Wr[((size_t)dir * 64 + (k - 64)) * NG + n];
        sW[n * 128 + k] = (_Float16)v;
    }
    for (int i = tid; i < 2 * DMAX * U_; i += THREADS) {
        sH[i] = (_Float16)0.f;
        sC[i] = 0.f;
    }

    // ---- preload x for diagonal 0 into x-buffer 0 ----
    for (int s = tid; s < DMAX * (U_ / 4); s += THREADS) {
        const int k = s >> 4;
        if (k < 1) {   // diagonal 0 has one cell: (0,0)
            const int c4 = (s & 15) * 4;
            const int rr = fh ? (H_ - 1) : 0;
            const int cc = fw ? (W_ - 1) : 0;
            copy16_to_lds(x + (((size_t)bb * H_ + rr) * W_ + cc) * C_ + c4,
                          sX + k * U_ + c4);
        }
    }
    wait_async_copies();
    __syncthreads();

    const int wave   = tid >> 5;
    const int lane   = tid & 31;
    const int ln     = lane & 15;
    const int hiHalf = lane >> 4;        // 0 or 1
    const int mTile  = wave & 7;         // 8 M tiles cover 128 diag rows
    const int uGroup = wave >> 3;        // 2 groups of 2 u-tiles x 4 gates

    int pb = 0;                          // x double-buffer parity

    for (int d = 0; d < NDIAG; ++d) {
        const int i0  = max(0, d - (W_ - 1));
        const int i1  = min(H_ - 1, d);
        const int len = i1 - i0 + 1;
        // 2-way state rotation: read d-1 (p1) and d-2 (wr), write d into wr.
        const int p1 = (d + 1) & 1;
        const int wr = d & 1;
        const _Float16* hp1 = sH + p1 * DMAX * U_;
        const _Float16* hp2 = sH + wr * DMAX * U_;
        const float*    cp1 = sC + p1 * DMAX * U_;
        const float*    cp2 = sC + wr * DMAX * U_;
        const int i0p  = max(0, d - W_);        // diag d-1 row origin
        const int i0pp = max(0, d - 1 - W_);    // diag d-2 row origin

        // ---- stage: average h/c neighbors (4-wide vectorized) ----
        for (int i = tid; i < DMAX * (U_ / 4); i += THREADS) {
            const int k  = i >> 4;
            const int c4 = (i & 15) * 4;
            f4v hs = {0.f, 0.f, 0.f, 0.f};
            f4v cs = {0.f, 0.f, 0.f, 0.f};
            float cnt = 0.f;
            if (k < len) {
                const int r = i0 + k, c = d - r;
                if (r - 1 >= 0) {                 // up neighbor on diag d-1
                    const int s = (r - 1) - i0p;
                    h4v hv = *(const h4v*)(hp1 + s * U_ + c4);
                    cs += *(const f4v*)(cp1 + s * U_ + c4);
                    hs += (f4v){(float)hv[0], (float)hv[1], (float)hv[2], (float)hv[3]};
                    cnt += 1.f;
                }
                if (c - 1 >= 0) {                 // left neighbor on diag d-1
                    const int s = r - i0p;
                    h4v hv = *(const h4v*)(hp1 + s * U_ + c4);
                    cs += *(const f4v*)(cp1 + s * U_ + c4);
                    hs += (f4v){(float)hv[0], (float)hv[1], (float)hv[2], (float)hv[3]};
                    cnt += 1.f;
                }
                if (r - 1 >= 0 && c - 1 >= 0) {   // up-left neighbor on diag d-2
                    const int s = (r - 1) - i0pp;
                    h4v hv = *(const h4v*)(hp2 + s * U_ + c4);
                    cs += *(const f4v*)(cp2 + s * U_ + c4);
                    hs += (f4v){(float)hv[0], (float)hv[1], (float)hv[2], (float)hv[3]};
                    cnt += 1.f;
                }
            }
            const float inv = (cnt > 0.f) ? (1.f / cnt) : 1.f;  // cnt=max(cnt,1)
            h4v ho;
            ho[0] = (_Float16)(hs[0] * inv); ho[1] = (_Float16)(hs[1] * inv);
            ho[2] = (_Float16)(hs[2] * inv); ho[3] = (_Float16)(hs[3] * inv);
            *(h4v*)(sHA + k * U_ + c4) = ho;
            *(f4v*)(sCP + k * U_ + c4) = cs * inv;
        }
        __syncthreads();

        // ---- kick off async x staging for diagonal d+1 (overlaps WMMA) ----
        // Rows >= len are left stale: their Z rows only feed masked cells.
        if (d + 1 < NDIAG) {
            const int i0n  = max(0, d + 1 - (W_ - 1));
            const int lenN = min(H_ - 1, d + 1) - i0n + 1;
            float* xNext = sX + (1 - pb) * DMAX * U_;
            for (int s = tid; s < DMAX * (U_ / 4); s += THREADS) {
                const int k = s >> 4;
                if (k < lenN) {
                    const int c4 = (s & 15) * 4;
                    const int r = i0n + k, c = d + 1 - r;
                    const int rr = fh ? (H_ - 1 - r) : r;
                    const int cc = fw ? (W_ - 1 - c) : c;
                    copy16_to_lds(x + (((size_t)bb * H_ + rr) * W_ + cc) * C_ + c4,
                                  xNext + k * U_ + c4);
                }
            }
        }

        // ---- WMMA: Z = [x|h] @ WcatT + b, tiles 16x16, K=128 in 4 steps ----
        const float* xCur = sX + pb * DMAX * U_;
        v8f acc[4][2];
#pragma unroll
        for (int g = 0; g < 4; ++g)
#pragma unroll
            for (int t = 0; t < 2; ++t) {
                const int nBase = (g * 4 + uGroup * 2 + t) * 16;
                const float bv = sB[nBase + ln];
                v8f a;
#pragma unroll
                for (int j = 0; j < 8; ++j) a[j] = bv;
                acc[g][t] = a;
            }

#pragma unroll
        for (int ks = 0; ks < 4; ++ks) {
            // A fragment: 16-bit A 16x32 layout.
            // lanes 0-15 hold K chunks {0-7,16-23}, lanes 16-31 {8-15,24-31}.
            const int koff = (ks & 1) * 32 + hiHalf * 8;
            const int mrow = mTile * 16 + ln;
            v16h af;
            if (ks < 2) {    // x half of A: f32 in LDS, convert at frag build
                const float* ap = xCur + mrow * U_ + koff;
#pragma unroll
                for (int v = 0; v < 4; ++v) {
                    af[2 * v]     = (_Float16)ap[2 * v];
                    af[2 * v + 1] = (_Float16)ap[2 * v + 1];
                }
#pragma unroll
                for (int v = 4; v < 8; ++v) {
                    af[2 * v]     = (_Float16)ap[16 + 2 * (v - 4)];
                    af[2 * v + 1] = (_Float16)ap[16 + 2 * (v - 4) + 1];
                }
            } else {         // h half of A: already f16
                const _Float16* ap = sHA + mrow * U_ + koff;
#pragma unroll
                for (int v = 0; v < 4; ++v) {
                    af[2 * v]     = ap[2 * v];
                    af[2 * v + 1] = ap[2 * v + 1];
                }
#pragma unroll
                for (int v = 4; v < 8; ++v) {
                    af[2 * v]     = ap[16 + 2 * (v - 4)];
                    af[2 * v + 1] = ap[16 + 2 * (v - 4) + 1];
                }
            }
            const int kBase = ks * 32;
#pragma unroll
            for (int g = 0; g < 4; ++g)
#pragma unroll
                for (int t = 0; t < 2; ++t) {
                    // B fragment: 16-bit B 32x16; lane holds N=nBase+ln,
                    // K = kBase + hiHalf*8 + {0-7,16-23} (k-contiguous in sW).
                    const int nTile = g * 4 + uGroup * 2 + t;
                    const _Float16* bp =
                        sW + (nTile * 16 + ln) * 128 + kBase + hiHalf * 8;
                    v16h bf;
#pragma unroll
                    for (int e = 0; e < 16; ++e) bf[e] = bp[e < 8 ? e : e + 8];
                    acc[g][t] = __builtin_amdgcn_wmma_f32_16x16x32_f16(
                        false, af, false, bf, (short)0, acc[g][t], false, false);
                }
        }

        // ---- lane-local LSTM gating; update state; write output ----
        // C/D layout: element j -> M = mTile*16 + j + hiHalf*8, N = nTile*16 + ln.
        // Gates g=0..3 of unit u sit in acc[g][t] at the SAME lane/element.
        _Float16* hOut = sH + wr * DMAX * U_;
        float*    cOut = sC + wr * DMAX * U_;
#pragma unroll
        for (int t = 0; t < 2; ++t) {
            const int u = (uGroup * 2 + t) * 16 + ln;
#pragma unroll
            for (int j = 0; j < 8; ++j) {
                const int m = mTile * 16 + j + hiHalf * 8;
                const float zi = acc[0][t][j], zf = acc[1][t][j];
                const float zg = acc[2][t][j], zo = acc[3][t][j];
                float hn = 0.f, cn = 0.f;
                if (m < len) {
                    const float cprev = sCP[m * U_ + u];
                    const float si = 1.f / (1.f + __expf(-zi));
                    const float sf = 1.f / (1.f + __expf(-zf));
                    const float so = 1.f / (1.f + __expf(-zo));
                    cn = sf * cprev + si * tanhf(zg);
                    hn = so * tanhf(cn);
                    const int r = i0 + m, c = d - r;
                    const int rr = fh ? (H_ - 1 - r) : r;
                    const int cc = fw ? (W_ - 1 - c) : c;
                    out[(((size_t)bb * H_ + rr) * W_ + cc) * NG + dir * U_ + u] = hn;
                }
                hOut[m * U_ + u] = (_Float16)hn;
                cOut[m * U_ + u] = cn;
            }
        }

        wait_async_copies();   // next diagonal's x resident in LDS
        __syncthreads();       // state + x published before next iteration
        pb ^= 1;
    }
}

extern "C" void kernel_launch(void* const* d_in, const int* in_sizes, int n_in,
                              void* d_out, int out_size, void* d_ws, size_t ws_size,
                              hipStream_t stream) {
    const float* x  = (const float*)d_in[0];
    const float* Wk = (const float*)d_in[1];
    const float* Wr = (const float*)d_in[2];
    const float* b  = (const float*)d_in[3];
    float* out = (float*)d_out;

    dim3 grid(16);          // 4 directions x 4 batches
    dim3 block(THREADS);    // 16 wave32 waves
    hipLaunchKernelGGL(mdlstm_scan_kernel, grid, block, SMEM_BYTES, stream,
                       x, Wk, Wr, b, out);
}
